// GraphEmbeddingModel_13314398617880
// MI455X (gfx1250) — matrix operand order repaired and checked by reading, compiled, not run on hardware
//
#include <hip/hip_runtime.h>

// HANConv inference pipeline for MI455X (gfx1250, wave32, WMMA).
// GEMM stages use v_wmma_f32_16x16x32_f16 (f16 in, f32 accumulate), one wave
// per 16x64 output strip. Weight matrices are pre-packed once into f16
// WMMA-fragment layout so B fragments are two b128 loads (no strided dword
// loads / cvts in the hot loop). A fragments use float4 (b128) loads.
// Edge-softmax/aggregation stages are bandwidth-bound scatter with native
// f32 atomics (the dominant ~2.6 GB of traffic; GEMMs are ~36 GF).

#define NREV  50000
#define NAUTH 50000
#define NEDGE 320000
#define CIN   128
#define CHID  256
#define COUT  128
#define NHEAD 8
#define NA    400000   // N * NHEAD for either node type
#define NTILE 4        // column tiles per wave (16x64 strip)

typedef __attribute__((ext_vector_type(16))) _Float16 v16h;
typedef __attribute__((ext_vector_type(8)))  float    v8f;

// ---------- order-preserving f32 <-> u32 encoding for atomic max ----------
__device__ __forceinline__ unsigned fenc(float x) {
  unsigned u = __float_as_uint(x);
  return (u & 0x80000000u) ? ~u : (u | 0x80000000u);
}
__device__ __forceinline__ float fdec(unsigned u) {
  u = (u & 0x80000000u) ? (u & 0x7fffffffu) : ~u;
  return __uint_as_float(u);
}

// ---------- weight pre-pack: B[K,Nc] f32 -> f16 fragments ----------
// Fragment layout (ISA 7.12): B 32x16 tile, col = lane&15, lanes 0-15 hold
// K={0..7,16..23}, lanes 16-31 hold K={8..15,24..31}. Stored as
// P[(kt*nct + ct)*32 + lane][16] so a fragment is one 32-byte vector load.
__global__ void pack_b(const float* __restrict__ B, _Float16* __restrict__ P,
                       int K, int Nc) {
  const int idx = blockIdx.x * blockDim.x + threadIdx.x;
  const int nkt = K >> 5, nct = Nc >> 4;
  if (idx >= nkt * nct * 32) return;
  const int lane = idx & 31;
  const int t    = idx >> 5;          // kt * nct + ct
  const int ct   = t % nct;
  const int kt   = t / nct;
  const int col  = ct * 16 + (lane & 15);
  const int kb   = kt * 32 + ((lane >> 4) << 3);
  _Float16* o = P + (size_t)idx * 16;
#pragma unroll
  for (int j = 0; j < 16; ++j) o[j] = (_Float16)B[(size_t)(kb + j + ((j >> 3) << 3)) * Nc + col];
}

__device__ __forceinline__ v16h load_b_packed(const _Float16* __restrict__ P,
                                              int kt, int ct, int nct, int lane) {
  return *(const v16h*)(P + ((((size_t)kt * nct + ct) * 32) + lane) * 16);
}

// ---------- A fragment loaders (f32 memory -> f16 fragment, b128 loads) ----
// A (16x32, row-major source): lane<16 -> row=lane, K slots {0..7, 16..23};
// lane>=16 -> row=lane-16, K slots {8..15, 24..31}. (ISA 7.12.2)
__device__ __forceinline__ v16h cvt16(float4 x0, float4 x1, float4 x2, float4 x3) {
  v16h a;
  a[0]  = (_Float16)x0.x; a[1]  = (_Float16)x0.y; a[2]  = (_Float16)x0.z; a[3]  = (_Float16)x0.w;
  a[4]  = (_Float16)x1.x; a[5]  = (_Float16)x1.y; a[6]  = (_Float16)x1.z; a[7]  = (_Float16)x1.w;
  a[8]  = (_Float16)x2.x; a[9]  = (_Float16)x2.y; a[10] = (_Float16)x2.z; a[11] = (_Float16)x2.w;
  a[12] = (_Float16)x3.x; a[13] = (_Float16)x3.y; a[14] = (_Float16)x3.z; a[15] = (_Float16)x3.w;
  return a;
}

__device__ __forceinline__ v16h load_a_frag(const float* __restrict__ A, int lda,
                                            int row0, int k0, int lane) {
  const int row = row0 + (lane & 15);
  const int kb  = k0 + ((lane >> 4) << 3);
  const float* p = A + (size_t)row * lda + kb;
  const float4 x0 = *(const float4*)(p + 0);
  const float4 x1 = *(const float4*)(p + 4);
  const float4 x2 = *(const float4*)(p + 16);
  const float4 x3 = *(const float4*)(p + 20);
  return cvt16(x0, x1, x2, x3);
}

// A fragment built on the fly as w0*O0 + w1*O1 (semantic attention combine)
__device__ __forceinline__ v16h load_a_frag2(const float* __restrict__ O0,
                                             const float* __restrict__ O1, int lda,
                                             int row0, int k0, int lane,
                                             float w0, float w1) {
  const int row = row0 + (lane & 15);
  const int kb  = k0 + ((lane >> 4) << 3);
  const size_t base = (size_t)row * lda + kb;
  const float* p0 = O0 + base;
  const float* p1 = O1 + base;
  float4 a0 = *(const float4*)(p0 + 0),  b0 = *(const float4*)(p1 + 0);
  float4 a1 = *(const float4*)(p0 + 4),  b1 = *(const float4*)(p1 + 4);
  float4 a2 = *(const float4*)(p0 + 16), b2 = *(const float4*)(p1 + 16);
  float4 a3 = *(const float4*)(p0 + 20), b3 = *(const float4*)(p1 + 20);
  float4 x0 = make_float4(w0*a0.x + w1*b0.x, w0*a0.y + w1*b0.y, w0*a0.z + w1*b0.z, w0*a0.w + w1*b0.w);
  float4 x1 = make_float4(w0*a1.x + w1*b1.x, w0*a1.y + w1*b1.y, w0*a1.z + w1*b1.z, w0*a1.w + w1*b1.w);
  float4 x2 = make_float4(w0*a2.x + w1*b2.x, w0*a2.y + w1*b2.y, w0*a2.z + w1*b2.z, w0*a2.w + w1*b2.w);
  float4 x3 = make_float4(w0*a3.x + w1*b3.x, w0*a3.y + w1*b3.y, w0*a3.z + w1*b3.z, w0*a3.w + w1*b3.w);
  return cvt16(x0, x1, x2, x3);
}

// ---------- GEMM kernels (one wave = one 16x64 output strip) ----------
// C = A @ B + bias   (B pre-packed)
__global__ void gemm_proj(const float* __restrict__ A, const _Float16* __restrict__ Bp,
                          const float* __restrict__ bias, float* __restrict__ C,
                          int M, int K, int Nc) {
  const int lane = threadIdx.x & 31;
  const int row0 = blockIdx.x * 16;
  const int col0 = blockIdx.y * (16 * NTILE);
  const int nct  = Nc >> 4;
  const int ct0  = col0 >> 4;
  v8f acc[NTILE] = {};
  for (int k0 = 0; k0 < K; k0 += 32) {
    const v16h a = load_a_frag(A, K, row0, k0, lane);
    const int kt = k0 >> 5;
#pragma unroll
    for (int nt = 0; nt < NTILE; ++nt) {
      const v16h b = load_b_packed(Bp, kt, ct0 + nt, nct, lane);
      acc[nt] = __builtin_amdgcn_wmma_f32_16x16x32_f16(false, a, false, b,
                                                       (short)0, acc[nt], false, false);
    }
  }
  const int rbase = row0 + ((lane >> 4) << 3);
#pragma unroll
  for (int nt = 0; nt < NTILE; ++nt) {
    const int col  = col0 + nt * 16 + (lane & 15);
    const float bv = bias[col];
#pragma unroll
    for (int r = 0; r < 8; ++r) C[(size_t)(rbase + r) * Nc + col] = acc[nt][r] + bv;
  }
}

// colsum[col] += sum_rows tanh((A@B)[row,col] + bias[col])   (no C written)
__global__ void gemm_tanh_colsum(const float* __restrict__ A, const _Float16* __restrict__ Bp,
                                 const float* __restrict__ bias, float* __restrict__ colsum,
                                 int M, int K, int Nc) {
  const int lane = threadIdx.x & 31;
  const int row0 = blockIdx.x * 16;
  const int col0 = blockIdx.y * (16 * NTILE);
  const int nct  = Nc >> 4;
  const int ct0  = col0 >> 4;
  v8f acc[NTILE] = {};
  for (int k0 = 0; k0 < K; k0 += 32) {
    const v16h a = load_a_frag(A, K, row0, k0, lane);
    const int kt = k0 >> 5;
#pragma unroll
    for (int nt = 0; nt < NTILE; ++nt) {
      const v16h b = load_b_packed(Bp, kt, ct0 + nt, nct, lane);
      acc[nt] = __builtin_amdgcn_wmma_f32_16x16x32_f16(false, a, false, b,
                                                       (short)0, acc[nt], false, false);
    }
  }
#pragma unroll
  for (int nt = 0; nt < NTILE; ++nt) {
    const int col  = col0 + nt * 16 + (lane & 15);
    const float bv = bias[col];
    float s = 0.f;
#pragma unroll
    for (int r = 0; r < 8; ++r) s += tanhf(acc[nt][r] + bv);
    atomicAdd(&colsum[col], s);
  }
}

// C = (attn[i0]*O0 + attn[i1]*O1) @ B + bias   (B pre-packed)
__global__ void gemm_final(const float* __restrict__ O0, const float* __restrict__ O1,
                           const float* __restrict__ attn, int i0, int i1,
                           const _Float16* __restrict__ Bp, const float* __restrict__ bias,
                           float* __restrict__ C, int M, int K, int Nc) {
  const int lane = threadIdx.x & 31;
  const int row0 = blockIdx.x * 16;
  const int col0 = blockIdx.y * (16 * NTILE);
  const int nct  = Nc >> 4;
  const int ct0  = col0 >> 4;
  const float w0 = attn[i0], w1 = attn[i1];
  v8f acc[NTILE] = {};
  for (int k0 = 0; k0 < K; k0 += 32) {
    const v16h a = load_a_frag2(O0, O1, K, row0, k0, lane, w0, w1);
    const int kt = k0 >> 5;
#pragma unroll
    for (int nt = 0; nt < NTILE; ++nt) {
      const v16h b = load_b_packed(Bp, kt, ct0 + nt, nct, lane);
      acc[nt] = __builtin_amdgcn_wmma_f32_16x16x32_f16(false, a, false, b,
                                                       (short)0, acc[nt], false, false);
    }
  }
  const int rbase = row0 + ((lane >> 4) << 3);
#pragma unroll
  for (int nt = 0; nt < NTILE; ++nt) {
    const int col  = col0 + nt * 16 + (lane & 15);
    const float bv = bias[col];
#pragma unroll
    for (int r = 0; r < 8; ++r) C[(size_t)(rbase + r) * Nc + col] = acc[nt][r] + bv;
  }
}

// ---------- node-level attention logits: alpha[n,h] = sum_d h[n,h,d]*att[h,d]
__global__ void alpha_kernel(const float* __restrict__ h, const float* __restrict__ att,
                             float* __restrict__ out, int N) {
  const int i = blockIdx.x * blockDim.x + threadIdx.x;
  if (i >= N * NHEAD) return;
  const int n = i >> 3, hd = i & 7;
  const float* hp = h + (size_t)n * CHID + hd * 32;
  const float* ap = att + hd * 32;
  float s = 0.f;
#pragma unroll
  for (int d = 0; d < 32; ++d) s += hp[d] * ap[d];
  out[i] = s;
}

// ---------- edge passes ----------
// pass 1: alpha = leaky_relu(asrc[src]+adst[dst]); store; atomic segment max
__global__ void edge_alpha_max(const int* __restrict__ src, const int* __restrict__ dst,
                               const float* __restrict__ asrc, const float* __restrict__ adst,
                               float* __restrict__ abuf, unsigned* __restrict__ amax) {
  const int i = blockIdx.x * blockDim.x + threadIdx.x;
  if (i >= NEDGE * NHEAD) return;
  const int e = i >> 3, hd = i & 7;
  const int s = src[e], d = dst[e];
  float a = asrc[s * NHEAD + hd] + adst[d * NHEAD + hd];
  a = (a > 0.f) ? a : 0.2f * a;
  abuf[i] = a;
  atomicMax(&amax[d * NHEAD + hd], fenc(a));
}

// pass 2: ex = exp(alpha - amax[dst]); store in place; atomic segment sum
__global__ void edge_expsum(const int* __restrict__ dst, float* __restrict__ abuf,
                            const unsigned* __restrict__ amax, float* __restrict__ denom) {
  const int i = blockIdx.x * blockDim.x + threadIdx.x;
  if (i >= NEDGE * NHEAD) return;
  const int e = i >> 3, hd = i & 7;
  const int d = dst[e];
  const float m  = fdec(amax[d * NHEAD + hd]);
  const float ex = __expf(abuf[i] - m);
  abuf[i] = ex;
  atomicAdd(&denom[d * NHEAD + hd], ex);
}

// pass 3: out[dst,:] += h_src[src,:] * (ex/denom[dst])   (64 lanes * float4 / edge)
__global__ void edge_aggregate(const int* __restrict__ src, const int* __restrict__ dst,
                               const float* __restrict__ hsrc, const float* __restrict__ exb,
                               const float* __restrict__ denom, float* __restrict__ out) {
  const int g = threadIdx.x >> 6;   // edge slot within block (0..3)
  const int t = threadIdx.x & 63;   // float4 index (0..63)
  const int e = blockIdx.x * 4 + g;
  if (e >= NEDGE) return;
  const int s = src[e], d = dst[e];
  const int hd = t >> 3;            // head for floats [4t..4t+3]
  const float w = exb[e * NHEAD + hd] / (denom[d * NHEAD + hd] + 1e-16f);
  const float4 v = reinterpret_cast<const float4*>(hsrc + (size_t)s * CHID)[t];
  float* op = out + (size_t)d * CHID + t * 4;
  atomicAdd(op + 0, v.x * w);
  atomicAdd(op + 1, v.y * w);
  atomicAdd(op + 2, v.z * w);
  atomicAdd(op + 3, v.w * w);
}

// ---------- utility kernels ----------
__global__ void fill_f32(float* __restrict__ p, float v, size_t n) {
  size_t i = (size_t)blockIdx.x * blockDim.x + threadIdx.x;
  const size_t st = (size_t)gridDim.x * blockDim.x;
  for (; i < n; i += st) p[i] = v;
}
__global__ void fill_u32(unsigned* __restrict__ p, unsigned v, size_t n) {
  size_t i = (size_t)blockIdx.x * blockDim.x + threadIdx.x;
  const size_t st = (size_t)gridDim.x * blockDim.x;
  for (; i < n; i += st) p[i] = v;
}
__global__ void relu_ip(float* __restrict__ p, size_t n) {
  size_t i = (size_t)blockIdx.x * blockDim.x + threadIdx.x;
  const size_t st = (size_t)gridDim.x * blockDim.x;
  for (; i < n; i += st) p[i] = fmaxf(p[i], 0.f);
}

// semantic-level attention: scores + softmax over the 2 metapaths per group
// colsum layout: [ar, rr, ra, aa] blocks of CHID
__global__ void semantic_softmax(const float* __restrict__ q,
                                 const float* __restrict__ colsum,
                                 float* __restrict__ attn) {
  if (threadIdx.x != 0 || blockIdx.x != 0) return;
  float s[4];
  for (int m = 0; m < 4; ++m) {
    float acc = 0.f;
    for (int f = 0; f < CHID; ++f) acc += q[f] * colsum[m * CHID + f];
    s[m] = acc;
  }
  s[0] *= (1.f / NREV);  s[1] *= (1.f / NREV);
  s[2] *= (1.f / NAUTH); s[3] *= (1.f / NAUTH);
  {
    const float m0 = fmaxf(s[0], s[1]);
    const float e0 = __expf(s[0] - m0), e1 = __expf(s[1] - m0);
    attn[0] = e0 / (e0 + e1); attn[1] = e1 / (e0 + e1);
  }
  {
    const float m0 = fmaxf(s[2], s[3]);
    const float e0 = __expf(s[2] - m0), e1 = __expf(s[3] - m0);
    attn[2] = e0 / (e0 + e1); attn[3] = e1 / (e0 + e1);
  }
}

extern "C" void kernel_launch(void* const* d_in, const int* in_sizes, int n_in,
                              void* d_out, int out_size, void* d_ws, size_t ws_size,
                              hipStream_t stream) {
  const float* x_rev   = (const float*)d_in[0];
  const float* x_auth  = (const float*)d_in[1];
  const int*   ei_ra   = (const int*)d_in[2];
  const int*   ei_ar   = (const int*)d_in[3];
  const int*   ei_aa   = (const int*)d_in[4];
  const int*   ei_rr   = (const int*)d_in[5];
  const float* W_rev   = (const float*)d_in[6];
  const float* b_rev   = (const float*)d_in[7];
  const float* W_auth  = (const float*)d_in[8];
  const float* b_auth  = (const float*)d_in[9];
  const float* att_src = (const float*)d_in[10];
  const float* att_dst = (const float*)d_in[11];
  const float* W_k     = (const float*)d_in[12];
  const float* b_k     = (const float*)d_in[13];
  const float* qv      = (const float*)d_in[14];
  const float* W_fc    = (const float*)d_in[15];
  const float* b_fc    = (const float*)d_in[16];
  float* out = (float*)d_out;
  float* W   = (float*)d_ws;

  // ---- workspace layout (floats) ----
  const size_t SZ_H    = (size_t)NREV * CHID;          // 12.8M
  const size_t o_h_rev  = 0;
  const size_t o_h_auth = SZ_H;
  const size_t o_out_ar = 2 * SZ_H;   // reviewer group: [ar, rr]
  const size_t o_out_rr = 3 * SZ_H;
  const size_t o_out_ra = 4 * SZ_H;   // author group:   [ra, aa]
  const size_t o_out_aa = 5 * SZ_H;
  const size_t o_alpha  = 6 * SZ_H;                    // 8 * NA
  const size_t o_amax   = o_alpha + 8 * (size_t)NA;    // 4 * NA (u32)
  const size_t o_denom  = o_amax + 4 * (size_t)NA;     // 4 * NA
  const size_t o_exbuf  = o_denom + 4 * (size_t)NA;    // NEDGE * 8
  const size_t o_colsum = o_exbuf + (size_t)NEDGE * NHEAD; // 4 * CHID
  const size_t o_attn   = o_colsum + 4 * CHID;         // 4
  // packed f16 weights, 64-byte aligned (sizes in halves)
  const size_t o_pack   = (o_attn + 4 + 15) & ~(size_t)15;
  _Float16* PB = (_Float16*)(W + o_pack);
  const size_t hp_rev  = 0;                         // (128/32)*(256/16)*512 = 32768
  const size_t hp_auth = 32768;
  const size_t hp_k    = 65536;                     // (256/32)*(256/16)*512 = 65536
  const size_t hp_fc   = 131072;                    // (256/32)*(128/16)*512 = 32768

  const dim3 b32(32);
  const dim3 b256(256);

  // ---- init accumulators ----
  fill_f32<<<2048, b256, 0, stream>>>(W + o_out_ar, 0.f, 4 * SZ_H);
  fill_f32<<<512, b256, 0, stream>>>(W + o_denom, 0.f, 4 * (size_t)NA);
  fill_f32<<<8, b256, 0, stream>>>(W + o_colsum, 0.f, 4 * CHID + 4);
  fill_u32<<<512, b256, 0, stream>>>((unsigned*)(W + o_amax), 0x007FFFFFu, 4 * (size_t)NA);

  // ---- pre-pack weight matrices into f16 WMMA fragment layout ----
  pack_b<<<8, b256, 0, stream>>>(W_rev,  PB + hp_rev,  CIN,  CHID);
  pack_b<<<8, b256, 0, stream>>>(W_auth, PB + hp_auth, CIN,  CHID);
  pack_b<<<16, b256, 0, stream>>>(W_k,   PB + hp_k,    CHID, CHID);
  pack_b<<<8, b256, 0, stream>>>(W_fc,   PB + hp_fc,   CHID, COUT);

  // ---- input projections: h = x @ W + b  (WMMA) ----
  gemm_proj<<<dim3(NREV / 16, CHID / (16 * NTILE)), b32, 0, stream>>>(
      x_rev, PB + hp_rev, b_rev, W + o_h_rev, NREV, CIN, CHID);
  gemm_proj<<<dim3(NAUTH / 16, CHID / (16 * NTILE)), b32, 0, stream>>>(
      x_auth, PB + hp_auth, b_auth, W + o_h_auth, NAUTH, CIN, CHID);

  // ---- node-level attention logits (8 arrays) ----
  const int agrid = (NA + 255) / 256;
  float* AL = W + o_alpha;
  alpha_kernel<<<agrid, b256, 0, stream>>>(W + o_h_rev,  att_src + 0 * CHID, AL + 0 * NA, NREV);   // as_ra
  alpha_kernel<<<agrid, b256, 0, stream>>>(W + o_h_auth, att_dst + 0 * CHID, AL + 1 * NA, NAUTH);  // ad_ra
  alpha_kernel<<<agrid, b256, 0, stream>>>(W + o_h_auth, att_src + 1 * CHID, AL + 2 * NA, NAUTH);  // as_ar
  alpha_kernel<<<agrid, b256, 0, stream>>>(W + o_h_rev,  att_dst + 1 * CHID, AL + 3 * NA, NREV);   // ad_ar
  alpha_kernel<<<agrid, b256, 0, stream>>>(W + o_h_auth, att_src + 2 * CHID, AL + 4 * NA, NAUTH);  // as_aa
  alpha_kernel<<<agrid, b256, 0, stream>>>(W + o_h_auth, att_dst + 2 * CHID, AL + 5 * NA, NAUTH);  // ad_aa
  alpha_kernel<<<agrid, b256, 0, stream>>>(W + o_h_rev,  att_src + 3 * CHID, AL + 6 * NA, NREV);   // as_rr
  alpha_kernel<<<agrid, b256, 0, stream>>>(W + o_h_rev,  att_dst + 3 * CHID, AL + 7 * NA, NREV);   // ad_rr

  // ---- per-edge-type segment softmax + aggregation ----
  const int* eis[4]    = { ei_ra, ei_ar, ei_aa, ei_rr };
  const size_t hsrcs[4]= { o_h_rev, o_h_auth, o_h_auth, o_h_rev };
  const size_t outs[4] = { o_out_ra, o_out_ar, o_out_aa, o_out_rr };
  const int egrid  = (NEDGE * NHEAD + 255) / 256;
  const int eggrid = (NEDGE + 3) / 4;
  for (int et = 0; et < 4; ++et) {
    const int* src = eis[et];
    const int* dst = eis[et] + NEDGE;
    const float* asrc = AL + (size_t)(2 * et + 0) * NA;
    const float* adst = AL + (size_t)(2 * et + 1) * NA;
    unsigned* amax = (unsigned*)(W + o_amax) + (size_t)et * NA;
    float* denom   = W + o_denom + (size_t)et * NA;
    float* exb     = W + o_exbuf;  // reused serially across edge types
    edge_alpha_max<<<egrid, b256, 0, stream>>>(src, dst, asrc, adst, exb, amax);
    edge_expsum<<<egrid, b256, 0, stream>>>(dst, exb, amax, denom);
    edge_aggregate<<<eggrid, b256, 0, stream>>>(src, dst, W + hsrcs[et], exb, denom, W + outs[et]);
  }

  // ---- ReLU on all four metapath outputs (contiguous block) ----
  relu_ip<<<2048, b256, 0, stream>>>(W + o_out_ar, 4 * SZ_H);

  // ---- semantic scores: colsum of tanh(out @ W_k + b_k)  (WMMA, fused) ----
  gemm_tanh_colsum<<<dim3(NREV / 16, CHID / (16 * NTILE)), b32, 0, stream>>>(
      W + o_out_ar, PB + hp_k, b_k, W + o_colsum + 0 * CHID, NREV, CHID, CHID);
  gemm_tanh_colsum<<<dim3(NREV / 16, CHID / (16 * NTILE)), b32, 0, stream>>>(
      W + o_out_rr, PB + hp_k, b_k, W + o_colsum + 1 * CHID, NREV, CHID, CHID);
  gemm_tanh_colsum<<<dim3(NAUTH / 16, CHID / (16 * NTILE)), b32, 0, stream>>>(
      W + o_out_ra, PB + hp_k, b_k, W + o_colsum + 2 * CHID, NAUTH, CHID, CHID);
  gemm_tanh_colsum<<<dim3(NAUTH / 16, CHID / (16 * NTILE)), b32, 0, stream>>>(
      W + o_out_aa, PB + hp_k, b_k, W + o_colsum + 3 * CHID, NAUTH, CHID, CHID);

  semantic_softmax<<<1, 32, 0, stream>>>(qv, W + o_colsum, W + o_attn);

  // ---- final: (attn-combined outs) @ W_fc + b_fc  (WMMA, fused combine) ----
  gemm_final<<<dim3(NREV / 16, COUT / (16 * NTILE)), b32, 0, stream>>>(
      W + o_out_ar, W + o_out_rr, W + o_attn, 0, 1, PB + hp_fc, b_fc,
      out, NREV, CHID, COUT);
  gemm_final<<<dim3(NAUTH / 16, COUT / (16 * NTILE)), b32, 0, stream>>>(
      W + o_out_ra, W + o_out_aa, W + o_attn, 2, 3, PB + hp_fc, b_fc,
      out + (size_t)NREV * COUT, NAUTH, CHID, COUT);
}
